// MulDecoder_6287832122075
// MI455X (gfx1250) — compile-verified, matrix operand order checked by reading, not run on hardware
//
#include <hip/hip_runtime.h>
#include <math.h>

typedef __attribute__((ext_vector_type(16))) _Float16 v16h;
typedef __attribute__((ext_vector_type(8)))  _Float16 v8h;
typedef __attribute__((ext_vector_type(2)))  _Float16 v2h;
typedef __attribute__((ext_vector_type(8)))  float    v8f;

#define CB 4
#define CT 512
#define CS 512
#define CH 512
#define CKD 512
#define CVD 512
#define CFF 2048
#define CNH 8
#define CEMB 300
#define CE 8
#define CDH 64

// k index of fragment element pair j (even k), lane half kh:
//   elements 0..7  -> K = 0..7   (+8*kh)
//   elements 8..15 -> K = 16..23 (+8*kh)
__device__ __forceinline__ int frag_k(int j, int kh) {
  return ((j < 4) ? (2 * j) : (16 + 2 * (j - 4))) + 8 * kh;
}

// Build a 16x32 f16 fragment from K-contiguous f16 LDS row: two b128 loads.
// p points at row_base + 8*kh (16B aligned); second chunk at +16 elements.
__device__ __forceinline__ v16h ldfrag(const _Float16* p) {
  v8h lo = *(const v8h*)p;
  v8h hi = *(const v8h*)(p + 16);
  return __builtin_shufflevector(lo, hi, 0, 1, 2, 3, 4, 5, 6, 7, 8, 9, 10, 11,
                                 12, 13, 14, 15);
}

// ---------------------------------------------------------------------------
// WMMA GEMM: C[M,N] = act( scale * sum_tap A_shift @ B + bias ) + res
//   A: [M,K] f32 row-major (rows shifted per tap for causal conv, zero-pad)
//   B: [K,N] f32 row-major  (or [N,K] when TRANSB, i.e. C = A @ B^T)
// Block: 256 threads / 8 waves, 128x64 tile; wave -> 32x32 (4 WMMA / k-step).
// LDS double-buffered f16 tiles; next tile register-staged during WMMAs.
// ---------------------------------------------------------------------------
template <int NTAPS, bool TRANSB, bool BIAS, bool RELU, bool RES>
__global__ __launch_bounds__(256) void gemm_wmma(
    const float* __restrict__ A, const float* __restrict__ Bw,
    const float* __restrict__ bias, const float* __restrict__ res,
    float* __restrict__ C, int M, int N, int K, int Tmod, float outScale,
    long long sA, long long sB, long long sC) {
  A += (size_t)blockIdx.z * sA;
  Bw += (size_t)blockIdx.z * sB;
  C += (size_t)blockIdx.z * sC;
  const float* resp = RES ? (res + (size_t)blockIdx.z * sC) : nullptr;

  __shared__ alignas(16) _Float16 As[2][128 * 40];  // [m][k] f16, stride 40
  __shared__ alignas(16) _Float16 Bs[2][64 * 40];   // [n][k] f16, stride 40

  const int tid = threadIdx.x;
  const int lane = tid & 31;
  const int wave = tid >> 5;
  const int wy = wave & 3;   // 32-row sub-tile
  const int wx = wave >> 2;  // 32-col slab
  const int m0 = blockIdx.y * 128;
  const int n0 = blockIdx.x * 64;
  const int n16 = lane & 15;
  const int kh = lane >> 4;

  const int kTiles = (K + 31) >> 5;
  const int iters = NTAPS * kTiles;

  float2 aR[8];
  float2 bR[4];

  auto loadRegs = [&](int it) {
    int tap = (NTAPS == 3) ? (it / kTiles) : 0;
    int kt = (NTAPS == 3) ? (it - tap * kTiles) : it;
    int kb = kt << 5;
    int shift = (NTAPS == 3) ? (2 - tap) : 0;
    const float* Bt = Bw + (size_t)tap * (size_t)K * (size_t)N;
#pragma unroll
    for (int i = 0; i < 8; ++i) {  // A tile 128x32 = 2048 pairs-of-2? (1024x2)
      int pidx = tid + i * 256;    // 2048 pairs total, pair = 2 consecutive k
      int kp = pidx & 15;
      int r = pidx >> 4;
      int gr = m0 + r;
      int k = kb + 2 * kp;
      bool ok = (k < K);
      int srow = gr;
      if (NTAPS == 3) {
        ok = ok && ((gr % Tmod) >= shift);
        srow = gr - shift;
      }
      if (ok) {
        aR[i] = *(const float2*)&A[(size_t)srow * K + k];
      } else {
        aR[i].x = 0.f;
        aR[i].y = 0.f;
      }
    }
#pragma unroll
    for (int i = 0; i < 4; ++i) {  // B tile 32x64 -> 1024 k-pairs
      int pidx = tid + i * 256;
      if (!TRANSB) {
        int n = pidx & 63;
        int kp = pidx >> 6;
        int k = kb + 2 * kp;
        if (k < K) {
          bR[i].x = Bt[(size_t)k * N + n0 + n];
          bR[i].y = Bt[(size_t)(k + 1) * N + n0 + n];
        } else {
          bR[i].x = 0.f;
          bR[i].y = 0.f;
        }
      } else {
        int kp = pidx & 15;
        int n = pidx >> 4;
        int k = kb + 2 * kp;
        if (k < K) {
          bR[i] = *(const float2*)&Bt[(size_t)(n0 + n) * K + k];
        } else {
          bR[i].x = 0.f;
          bR[i].y = 0.f;
        }
      }
    }
  };

  auto storeRegs = [&](int buf) {
#pragma unroll
    for (int i = 0; i < 8; ++i) {
      int pidx = tid + i * 256;
      int kp = pidx & 15;
      int r = pidx >> 4;
      v2h p = {(_Float16)aR[i].x, (_Float16)aR[i].y};
      *(v2h*)&As[buf][r * 40 + 2 * kp] = p;
    }
#pragma unroll
    for (int i = 0; i < 4; ++i) {
      int pidx = tid + i * 256;
      int n, kp;
      if (!TRANSB) {
        n = pidx & 63;
        kp = pidx >> 6;
      } else {
        kp = pidx & 15;
        n = pidx >> 4;
      }
      v2h p = {(_Float16)bR[i].x, (_Float16)bR[i].y};
      *(v2h*)&Bs[buf][n * 40 + 2 * kp] = p;
    }
  };

  loadRegs(0);
  storeRegs(0);
  __syncthreads();

  v8f c00 = {}, c01 = {}, c10 = {}, c11 = {};

  for (int it = 0; it < iters; ++it) {
    const int cur = it & 1;
    const bool hasNext = (it + 1 < iters);
    if (hasNext) loadRegs(it + 1);  // global loads overlap the WMMAs below

    v16h a0 = ldfrag(&As[cur][(wy * 32 + n16) * 40 + 8 * kh]);
    v16h a1 = ldfrag(&As[cur][(wy * 32 + 16 + n16) * 40 + 8 * kh]);
    v16h b0 = ldfrag(&Bs[cur][(wx * 32 + n16) * 40 + 8 * kh]);
    v16h b1 = ldfrag(&Bs[cur][(wx * 32 + 16 + n16) * 40 + 8 * kh]);
    c00 = __builtin_amdgcn_wmma_f32_16x16x32_f16(false, a0, false, b0, (short)0, c00, false, false);
    c01 = __builtin_amdgcn_wmma_f32_16x16x32_f16(false, a0, false, b1, (short)0, c01, false, false);
    c10 = __builtin_amdgcn_wmma_f32_16x16x32_f16(false, a1, false, b0, (short)0, c10, false, false);
    c11 = __builtin_amdgcn_wmma_f32_16x16x32_f16(false, a1, false, b1, (short)0, c11, false, false);

    if (hasNext) storeRegs(cur ^ 1);
    __syncthreads();
  }

  // epilogue: C layout lane = N col, VGPR j = M row (+8 for upper half-lanes)
  auto emit = [&](v8f& acc, int mt, int nt) {
#pragma unroll
    for (int j = 0; j < 8; ++j) {
      int m = mt + j + 8 * kh;
      int n = nt + n16;
      float v = acc[j] * outScale;
      if (BIAS) v += bias[n];
      if (RELU) v = fmaxf(v, 0.f);
      if (RES) v += resp[(size_t)m * N + n];
      C[(size_t)m * N + n] = v;
    }
  };
  const int mB = m0 + wy * 32;
  const int nB = n0 + wx * 32;
  emit(c00, mB, nB);
  emit(c01, mB, nB + 16);
  emit(c10, mB + 16, nB);
  emit(c11, mB + 16, nB + 16);
}

// ---------------------------------------------------------------------------
// Flash attention: one wave = 16 q rows of one (b, h). Online softmax in f32,
// P and transposed-V staged as f16 in per-wave LDS, PV via WMMA.
// ---------------------------------------------------------------------------
template <bool CAUSAL>
__global__ __launch_bounds__(128) void flash_attn(
    const float* __restrict__ Q, const float* __restrict__ Km,
    const float* __restrict__ Vm, float* __restrict__ O, int T, int S) {
  const int lane = threadIdx.x & 31;
  const int wave = threadIdx.x >> 5;
  const int b = blockIdx.z, h = blockIdx.y;
  const int qt0 = blockIdx.x * 64 + wave * 16;
  const int n16 = lane & 15;
  const int kh = lane >> 4;

  const float* qp = Q + ((size_t)b * T + qt0) * CKD + h * CDH;
  const float* kp = Km + ((size_t)b * S) * CKD + h * CDH;
  const float* vp = Vm + ((size_t)b * S) * CVD + h * CDH;

  __shared__ alignas(16) _Float16 Ps[4][16 * 40];  // P:  [m][s-chunk k]
  __shared__ alignas(16) _Float16 Vs[4][64 * 40];  // Vt: [dh][s-chunk k]
  _Float16* Pw = Ps[wave];
  _Float16* Vw = Vs[wave];

  // q fragments over dh = 64 (two k-chunks of 32)
  v16h aq0, aq1;
  {
    const float* r = qp + (size_t)n16 * CKD;
#pragma unroll
    for (int j = 0; j < 8; ++j) {
      int k0 = frag_k(j, kh);
      float2 x0 = *(const float2*)(r + k0);
      float2 x1 = *(const float2*)(r + 32 + k0);
      aq0[2 * j] = (_Float16)x0.x;
      aq0[2 * j + 1] = (_Float16)x0.y;
      aq1[2 * j] = (_Float16)x1.x;
      aq1[2 * j + 1] = (_Float16)x1.y;
    }
  }

  v8f o[4] = {v8f{}, v8f{}, v8f{}, v8f{}};
  float mrow[8], srow[8];
#pragma unroll
  for (int j = 0; j < 8; ++j) {
    mrow[j] = -1e30f;
    srow[j] = 0.f;
  }

  const int sEnd = CAUSAL ? (qt0 + 16) : S;
  for (int s0 = 0; s0 < sEnd; s0 += 32) {
    // ---- stage V chunk [32 s x 64 dh] transposed to Vw[dh][s] as f16 ----
#pragma unroll
    for (int i = 0; i < 16; ++i) {
      int u = lane + i * 32;
      int d2 = u & 31;   // dh pair index
      int sp = u >> 5;   // s pair index
      int s = 2 * sp;
      const float* vb = vp + (size_t)(s0 + s) * CVD + 2 * d2;
      float2 r0 = *(const float2*)vb;
      float2 r1 = *(const float2*)(vb + CVD);
      v2h q0 = {(_Float16)r0.x, (_Float16)r1.x};
      v2h q1 = {(_Float16)r0.y, (_Float16)r1.y};
      *(v2h*)&Vw[(2 * d2) * 40 + s] = q0;
      *(v2h*)&Vw[(2 * d2 + 1) * 40 + s] = q1;
    }

    // ---- K fragments for s-tiles [s0,s0+16), [s0+16,s0+32), dh split 0/32
    v16h kb00, kb01, kb10, kb11;
    {
      const float* r0 = kp + (size_t)(s0 + n16) * CKD;
      const float* r1 = kp + (size_t)(s0 + 16 + n16) * CKD;
#pragma unroll
      for (int j = 0; j < 8; ++j) {
        int k0 = frag_k(j, kh);
        float2 a0 = *(const float2*)(r0 + k0);
        float2 a1 = *(const float2*)(r0 + 32 + k0);
        float2 b0 = *(const float2*)(r1 + k0);
        float2 b1 = *(const float2*)(r1 + 32 + k0);
        kb00[2 * j] = (_Float16)a0.x; kb00[2 * j + 1] = (_Float16)a0.y;
        kb01[2 * j] = (_Float16)a1.x; kb01[2 * j + 1] = (_Float16)a1.y;
        kb10[2 * j] = (_Float16)b0.x; kb10[2 * j + 1] = (_Float16)b0.y;
        kb11[2 * j] = (_Float16)b1.x; kb11[2 * j + 1] = (_Float16)b1.y;
      }
    }
    v8f l0 = {};
    v8f l1 = {};
    l0 = __builtin_amdgcn_wmma_f32_16x16x32_f16(false, aq0, false, kb00, (short)0, l0, false, false);
    l0 = __builtin_amdgcn_wmma_f32_16x16x32_f16(false, aq1, false, kb01, (short)0, l0, false, false);
    l1 = __builtin_amdgcn_wmma_f32_16x16x32_f16(false, aq0, false, kb10, (short)0, l1, false, false);
    l1 = __builtin_amdgcn_wmma_f32_16x16x32_f16(false, aq1, false, kb11, (short)0, l1, false, false);

    // ---- online softmax; 16-lane row reductions via xor shuffles ----
#pragma unroll
    for (int j = 0; j < 8; ++j) {
      int trow = qt0 + j + 8 * kh;
      float e0 = l0[j], e1 = l1[j];
      if (CAUSAL) {
        if (s0 + n16 > trow) e0 = -1e30f;
        if (s0 + 16 + n16 > trow) e1 = -1e30f;
      }
      float cm = fmaxf(e0, e1);
      cm = fmaxf(cm, __shfl_xor(cm, 1));
      cm = fmaxf(cm, __shfl_xor(cm, 2));
      cm = fmaxf(cm, __shfl_xor(cm, 4));
      cm = fmaxf(cm, __shfl_xor(cm, 8));
      float nm = fmaxf(mrow[j], cm);
      float sc = __expf(mrow[j] - nm);
      float p0 = __expf(e0 - nm);
      float p1 = __expf(e1 - nm);
      float cs = p0 + p1;
      cs += __shfl_xor(cs, 1);
      cs += __shfl_xor(cs, 2);
      cs += __shfl_xor(cs, 4);
      cs += __shfl_xor(cs, 8);
      srow[j] = srow[j] * sc + cs;
      mrow[j] = nm;
      o[0][j] *= sc; o[1][j] *= sc; o[2][j] *= sc; o[3][j] *= sc;
      int m = j + 8 * kh;
      Pw[m * 40 + n16] = (_Float16)p0;       // C-layout -> A-layout transpose
      Pw[m * 40 + 16 + n16] = (_Float16)p1;  // via per-wave LDS patch
    }
    asm volatile("s_wait_dscnt 0" ::: "memory");  // wave-local LDS RAW fence

    v16h ap = ldfrag(&Pw[n16 * 40 + 8 * kh]);
#pragma unroll
    for (int d = 0; d < 4; ++d) {
      v16h bv = ldfrag(&Vw[(d * 16 + n16) * 40 + 8 * kh]);
      o[d] = __builtin_amdgcn_wmma_f32_16x16x32_f16(false, ap, false, bv,
                                                    (short)0, o[d], false, false);
    }
  }

  float* op = O + ((size_t)b * T + qt0) * CVD + h * CDH;
#pragma unroll
  for (int d = 0; d < 4; ++d)
#pragma unroll
    for (int j = 0; j < 8; ++j) {
      int m = j + 8 * kh;
      op[(size_t)m * CVD + d * 16 + n16] = o[d][j] / srow[j];
    }
}

// ---------------------------------------------------------------------------
// LayerNorm with unbiased std (ddof=1), eps added to std (not var)
// ---------------------------------------------------------------------------
__global__ __launch_bounds__(128) void ln_kernel(const float* __restrict__ X,
                                                 const float* __restrict__ g,
                                                 const float* __restrict__ bb,
                                                 float* __restrict__ Y, int Hn) {
  const int row = blockIdx.x;
  const float* x = X + (size_t)row * Hn;
  float* y = Y + (size_t)row * Hn;
  float s = 0.f, ss = 0.f;
  for (int i = threadIdx.x; i < Hn; i += blockDim.x) {
    float v = x[i];
    s += v;
    ss += v * v;
  }
#pragma unroll
  for (int m = 16; m >= 1; m >>= 1) {
    s += __shfl_xor(s, m);
    ss += __shfl_xor(ss, m);
  }
  __shared__ float red[2][4];
  int lane = threadIdx.x & 31, w = threadIdx.x >> 5;
  if (lane == 0) {
    red[0][w] = s;
    red[1][w] = ss;
  }
  __syncthreads();
  s = red[0][0] + red[0][1] + red[0][2] + red[0][3];
  ss = red[1][0] + red[1][1] + red[1][2] + red[1][3];
  float mu = s / Hn;
  float var = (ss - mu * s) / (float)(Hn - 1);
  float inv = 1.f / (sqrtf(fmaxf(var, 0.f)) + 1e-6f);
  for (int i = threadIdx.x; i < Hn; i += blockDim.x)
    y[i] = g[i] * (x[i] - mu) * inv + bb[i];
}

// x += timing_signal(t, c)
__global__ void add_timing(float* __restrict__ X, int total) {
  int idx = blockIdx.x * blockDim.x + threadIdx.x;
  if (idx >= total) return;
  int c = idx % CH;
  int t = (idx / CH) % CT;
  const int nts = CH / 2;
  int i = (c < nts) ? c : (c - nts);
  float inv = __expf(-(float)i * (logf(1.0e4f) / (float)(nts - 1)));
  float st = (float)t * inv;
  X[idx] += (c < nts) ? __sinf(st) : __cosf(st);
}

// dst += gates[b, e] * src   (gated expert combine)
__global__ void axpy_gate(float* __restrict__ dst, const float* __restrict__ src,
                          const float* __restrict__ gates, int e, int total) {
  int idx = blockIdx.x * blockDim.x + threadIdx.x;
  if (idx >= total) return;
  int bidx = idx / (CT * CH);
  dst[idx] += gates[bidx * CE + e] * src[idx];
}

// ---------------------------------------------------------------------------
// Host-side orchestration
// ---------------------------------------------------------------------------
struct LayerP {
  // sorted-key pytree order of the layer dict
  const float* p[18];
  // 0:b1 1:b2 2:ca_k 3:ca_o 4:ca_q 5:ca_v 6:ln1_b 7:ln1_g 8:ln2_b 9:ln2_g
  // 10:ln3_b 11:ln3_g 12:sa_k 13:sa_o 14:sa_q 15:sa_v 16:w1 17:w2
};

extern "C" void kernel_launch(void* const* d_in, const int* in_sizes, int n_in,
                              void* d_out, int out_size, void* d_ws,
                              size_t ws_size, hipStream_t stream) {
  (void)in_sizes; (void)n_in; (void)out_size; (void)ws_size;

  const float* inputs = (const float*)d_in[0];  // [B,T,EMB]
  const float* enc    = (const float*)d_in[1];  // [B,S,H]
  const float* gates  = (const float*)d_in[4];  // [B,E,1,1]
  const float* proj_w = (const float*)d_in[5];  // [EMB,H]
  const float* fin_g  = (const float*)d_in[60];
  const float* fin_b  = (const float*)d_in[61];

  static const size_t leafSz[18] = {
      CFF, CH, (size_t)CH * CKD, (size_t)CVD * CH, (size_t)CH * CKD,
      (size_t)CH * CVD, CH, CH, CH, CH, CH, CH,
      (size_t)CH * CKD, (size_t)CVD * CH, (size_t)CH * CKD, (size_t)CH * CVD,
      (size_t)3 * CH * CFF, (size_t)3 * CFF * CH};

  auto getLayer = [&](int base, int e) {
    LayerP L;
    for (int i = 0; i < 18; ++i)
      L.p[i] = (const float*)d_in[base + i] + (size_t)e * leafSz[i];
    return L;
  };

  const size_t BTH = (size_t)CB * CT * CH;   // 1M floats
  const size_t BTF = (size_t)CB * CT * CFF;  // 4M floats
  float* ws = (float*)d_ws;
  float* X    = ws; ws += BTH;
  float* XN   = ws; ws += BTH;
  float* Qb   = ws; ws += BTH;
  float* Kb   = ws; ws += BTH;
  float* Vb   = ws; ws += BTH;
  float* CTX  = ws; ws += BTH;
  float* X2   = ws; ws += BTH;
  float* X3   = ws; ws += BTH;
  float* LOUT = ws; ws += BTH;
  float* COMB = ws; ws += BTH;
  float* FFB  = ws; ws += BTF;

  const int BT = CB * CT;
  const dim3 gP(CH / 64, BT / 128, 1);    // N=512 GEMMs over 2048 rows
  const dim3 gF1(CFF / 64, BT / 128, 1);  // conv1 N=2048
  const dim3 gAtt(CT / 64, CNH, CB);      // flash attention
  const dim3 gLG(CS / 64, CT / 128, CB);  // per-batch logits GEMM
  const float qscale = 0.125f;            // dh^-0.5, dh=64

  auto run_layer = [&](const LayerP& L, const float* xin, float* xout,
                       float* attnOut) {
    // --- self attention ---
    ln_kernel<<<BT, 128, 0, stream>>>(xin, L.p[7], L.p[6], XN, CH);
    gemm_wmma<1, false, false, false, false><<<gP, 256, 0, stream>>>(
        XN, L.p[14], nullptr, nullptr, Qb, BT, CKD, CH, CT, qscale, 0, 0, 0);
    gemm_wmma<1, false, false, false, false><<<gP, 256, 0, stream>>>(
        XN, L.p[12], nullptr, nullptr, Kb, BT, CKD, CH, CT, 1.f, 0, 0, 0);
    gemm_wmma<1, false, false, false, false><<<gP, 256, 0, stream>>>(
        XN, L.p[15], nullptr, nullptr, Vb, BT, CVD, CH, CT, 1.f, 0, 0, 0);
    flash_attn<true><<<gAtt, 128, 0, stream>>>(Qb, Kb, Vb, CTX, CT, CT);
    gemm_wmma<1, false, false, false, true><<<gP, 256, 0, stream>>>(
        CTX, L.p[13], nullptr, xin, X2, BT, CH, CVD, CT, 1.f, 0, 0, 0);
    // --- cross attention ---
    ln_kernel<<<BT, 128, 0, stream>>>(X2, L.p[9], L.p[8], XN, CH);
    gemm_wmma<1, false, false, false, false><<<gP, 256, 0, stream>>>(
        XN, L.p[4], nullptr, nullptr, Qb, BT, CKD, CH, CT, qscale, 0, 0, 0);
    gemm_wmma<1, false, false, false, false><<<gP, 256, 0, stream>>>(
        enc, L.p[2], nullptr, nullptr, Kb, CB * CS, CKD, CH, CS, 1.f, 0, 0, 0);
    gemm_wmma<1, false, false, false, false><<<gP, 256, 0, stream>>>(
        enc, L.p[5], nullptr, nullptr, Vb, CB * CS, CVD, CH, CS, 1.f, 0, 0, 0);
    if (attnOut) {
      // head-summed pre-softmax logits: (Q @ K^T) / NH  (Q already * dh^-0.5)
      gemm_wmma<1, true, false, false, false><<<gLG, 256, 0, stream>>>(
          Qb, Kb, nullptr, nullptr, attnOut, CT, CS, CKD, CT, 1.f / CNH,
          (long long)CT * CKD, (long long)CS * CKD, (long long)CT * CS);
    }
    flash_attn<false><<<gAtt, 128, 0, stream>>>(Qb, Kb, Vb, CTX, CT, CS);
    gemm_wmma<1, false, false, false, true><<<gP, 256, 0, stream>>>(
        CTX, L.p[3], nullptr, X2, X3, BT, CH, CVD, CT, 1.f, 0, 0, 0);
    // --- causal conv FFN ---
    ln_kernel<<<BT, 128, 0, stream>>>(X3, L.p[11], L.p[10], XN, CH);
    gemm_wmma<3, false, true, true, false><<<gF1, 256, 0, stream>>>(
        XN, L.p[16], L.p[0], nullptr, FFB, BT, CFF, CH, CT, 1.f, 0, 0, 0);
    gemm_wmma<3, false, true, false, true><<<gP, 256, 0, stream>>>(
        FFB, L.p[17], L.p[1], X3, xout, BT, CH, CFF, CT, 1.f, 0, 0, 0);
  };

  // 1) embedding projection + timing signal
  gemm_wmma<1, false, false, false, false><<<gP, 256, 0, stream>>>(
      inputs, proj_w, nullptr, nullptr, X, BT, CH, CEMB, CT, 1.f, 0, 0, 0);
  add_timing<<<((int)BTH + 255) / 256, 256, 0, stream>>>(X, (int)BTH);

  // 2) basic layer (writes directly into COMB) + 8 gated experts
  run_layer(getLayer(6, 0), X, COMB, nullptr);
  for (int e = 0; e < CE; ++e) {
    run_layer(getLayer(24, e), X, LOUT, nullptr);
    axpy_gate<<<((int)BTH + 255) / 256, 256, 0, stream>>>(COMB, LOUT, gates, e,
                                                          (int)BTH);
  }

  // 3) final decoder layer (captures pre-softmax cross-attn logits)
  float* attnOut = (float*)d_out + BTH;
  run_layer(getLayer(42, 0), COMB, LOUT, attnOut);

  // 4) final LayerNorm into d_out[0 : B*T*H]
  ln_kernel<<<BT, 128, 0, stream>>>(LOUT, fin_g, fin_b, (float*)d_out, CH);
}
